// SkipGramHierarchicalSoftmax_64166811402274
// MI455X (gfx1250) — matrix operand order repaired and checked by reading, not compiled, not run on hardware
//
#include <hip/hip_runtime.h>

typedef __attribute__((ext_vector_type(16))) _Float16 v16h;
typedef __attribute__((ext_vector_type(8)))  _Float16 v8h;
typedef __attribute__((ext_vector_type(8)))  float    v8f;

#define EMBED 128
#define MAXP  20
#define WAVES 8      // waves per block (256 threads, wave32)
#define ROWS  24     // padded LDS row slots per wave (A-tile1 reads rows 16..23)

__global__ __launch_bounds__(256)
void hs_loss_kernel(const int* __restrict__ center,
                    const int* __restrict__ target,
                    const float* __restrict__ in_emb,
                    const float* __restrict__ inner_vec,
                    const int* __restrict__ paths,
                    const float* __restrict__ codes,
                    const float* __restrict__ masks,
                    float* __restrict__ block_part,
                    int nbatch)
{
    __shared__ __align__(16) _Float16 s_iv[WAVES][ROWS * EMBED]; // 48 KB
    __shared__ __align__(16) _Float16 s_h [WAVES][EMBED];        // 2 KB
    __shared__ float s_c[WAVES][MAXP];
    __shared__ float s_m[WAVES][MAXP];
    __shared__ float s_part[WAVES];

    const int lane = threadIdx.x & 31;
    const int wave = threadIdx.x >> 5;
    const int b    = blockIdx.x * WAVES + wave;

    float tot = 0.0f;
    if (b < nbatch) {
        const int ctr = center[b];
        const int tgt = target[b];

        // Stage center embedding h as f16: one coalesced 512B row per wave.
        {
            const float4 hv = *(const float4*)(in_emb + (size_t)ctr * EMBED + lane * 4);
            _Float16* dst = &s_h[wave][lane * 4];
            dst[0] = (_Float16)hv.x; dst[1] = (_Float16)hv.y;
            dst[2] = (_Float16)hv.z; dst[3] = (_Float16)hv.w;
        }

        // Per-level metadata (lanes 0..19), plus prefetch of gather rows.
        int p_reg = 0;
        if (lane < MAXP) {
            p_reg = paths[tgt * MAXP + lane];
            s_c[wave][lane] = codes[tgt * MAXP + lane];
            s_m[wave][lane] = masks[tgt * MAXP + lane];
            __builtin_prefetch(inner_vec + (size_t)p_reg * EMBED, 0, 1); // global_prefetch_b8
        }

        // Gather 20 inner-node rows into LDS (f32 -> f16).
        // Each row = 512B = exactly one wave-wide b128 load (32 lanes x 16B).
        #pragma unroll 4
        for (int l = 0; l < MAXP; ++l) {
            const int row = __shfl(p_reg, l);
            const float4 v = *(const float4*)(inner_vec + (size_t)row * EMBED + lane * 4);
            _Float16* dst = &s_iv[wave][l * EMBED + lane * 4];
            dst[0] = (_Float16)v.x; dst[1] = (_Float16)v.y;
            dst[2] = (_Float16)v.z; dst[3] = (_Float16)v.w;
        }

        // LDS region is wave-private: DS-counter ordering suffices (no barrier).
        asm volatile("s_wait_dscnt 0" ::: "memory");

        const int lane16 = lane & 15;
        const int hi     = lane >> 4;        // lane half select
        const int selA   = hi * 8;           // A: K {0..7,16..23} vs {8..15,24..31}
        const int selB   = hi * 8;           // B: K {0..7,16..23} vs {8..15,24..31}

        const _Float16* ivb = &s_iv[wave][0];
        const _Float16* hb  = &s_h[wave][0];
        const int row0 = lane16 * EMBED;                      // A-tile0: levels 0..15
        const int r1   = 16 + lane16;                         // A-tile1: levels 16..31
        const int row1 = (r1 < ROWS ? r1 : ROWS - 1) * EMBED; // clamp (rows >=20 unused)

        v8f acc0 = {}; v8f acc1 = {};
        #pragma unroll
        for (int kb = 0; kb < EMBED / 32; ++kb) {
            const int kh = kb * 32; // half-element base of this K-chunk
            v8h a0lo = *(const v8h*)(ivb + row0 + kh + selA);
            v8h a0hi = *(const v8h*)(ivb + row0 + kh + selA + 16);
            v8h a1lo = *(const v8h*)(ivb + row1 + kh + selA);
            v8h a1hi = *(const v8h*)(ivb + row1 + kh + selA + 16);
            v8h blo  = *(const v8h*)(hb + kh + selB);
            v8h bhi  = *(const v8h*)(hb + kh + selB + 16);
            v16h A0 = __builtin_shufflevector(a0lo, a0hi, 0,1,2,3,4,5,6,7,8,9,10,11,12,13,14,15);
            v16h A1 = __builtin_shufflevector(a1lo, a1hi, 0,1,2,3,4,5,6,7,8,9,10,11,12,13,14,15);
            v16h B  = __builtin_shufflevector(blo,  bhi,  0,1,2,3,4,5,6,7,8,9,10,11,12,13,14,15);
            acc0 = __builtin_amdgcn_wmma_f32_16x16x32_f16(false, A0, false, B,
                                                          (short)0, acc0, false, false);
            acc1 = __builtin_amdgcn_wmma_f32_16x16x32_f16(false, A1, false, B,
                                                          (short)0, acc1, false, false);
        }

        // C layout: VGPR r, lanes 0-15 -> M=r, lanes 16-31 -> M=r+8.
        // All N columns are identical (B broadcast); lanes 0 and 16 cover all levels.
        float part = 0.0f;
        const float* cw = s_c[wave];
        const float* mw = s_m[wave];
        #pragma unroll
        for (int r = 0; r < 8; ++r) {
            const int M = r + hi * 8;                 // levels 0..15
            const float z  = cw[M] * acc0[r];
            const float ls = fminf(z, 0.0f) - log1pf(__expf(-fabsf(z)));
            part += mw[M] * ls;
        }
        #pragma unroll
        for (int r = 0; r < 8; ++r) {
            const int M = 16 + r + hi * 8;            // levels 16..19 (hi==0, r<4)
            if (M < MAXP) {
                const float z  = cw[M] * acc1[r];
                const float ls = fminf(z, 0.0f) - log1pf(__expf(-fabsf(z)));
                part += mw[M] * ls;
            }
        }
        tot = -(__shfl(part, 0) + __shfl(part, 16));
    }

    if (lane == 0) s_part[wave] = tot;
    __syncthreads();
    if (threadIdx.x == 0) {
        float s = 0.0f;
        #pragma unroll
        for (int w = 0; w < WAVES; ++w) s += s_part[w];
        block_part[blockIdx.x] = s;
    }
}

__global__ __launch_bounds__(256)
void hs_reduce_kernel(const float* __restrict__ part, int n, float inv_batch,
                      float* __restrict__ out)
{
    __shared__ float red[256];
    float s = 0.0f;
    for (int i = threadIdx.x; i < n; i += 256) s += part[i];
    red[threadIdx.x] = s;
    __syncthreads();
    for (int off = 128; off > 0; off >>= 1) {
        if ((int)threadIdx.x < off) red[threadIdx.x] += red[threadIdx.x + off];
        __syncthreads();
    }
    if (threadIdx.x == 0) out[0] = red[0] * inv_batch;
}

extern "C" void kernel_launch(void* const* d_in, const int* in_sizes, int n_in,
                              void* d_out, int out_size, void* d_ws, size_t ws_size,
                              hipStream_t stream) {
    const int*   center    = (const int*)  d_in[0];
    const int*   target    = (const int*)  d_in[1];
    const float* in_emb    = (const float*)d_in[2];
    const float* inner_vec = (const float*)d_in[3];
    const int*   paths     = (const int*)  d_in[4];
    const float* codes     = (const float*)d_in[5];
    const float* masks     = (const float*)d_in[6];
    float* out  = (float*)d_out;
    float* part = (float*)d_ws;

    const int batch  = in_sizes[0];
    const int blocks = (batch + WAVES - 1) / WAVES;   // 4096 for BATCH=32768

    hs_loss_kernel<<<blocks, 256, 0, stream>>>(center, target, in_emb, inner_vec,
                                               paths, codes, masks, part, batch);
    hs_reduce_kernel<<<1, 256, 0, stream>>>(part, blocks, 1.0f / (float)batch, out);
}
